// WindowAttention_24927990186216
// MI455X (gfx1250) — matrix operand order repaired and checked by reading, not compiled, not run on hardware
//
#include <hip/hip_runtime.h>
#include <hip/hip_bf16.h>

typedef __attribute__((ext_vector_type(16))) __bf16 bf16x16;
typedef __attribute__((ext_vector_type(8)))  __bf16 bf16x8;
typedef __attribute__((ext_vector_type(2)))  __bf16 bf16x2;
typedef __attribute__((ext_vector_type(8)))  float  f32x8;

#define CDIM 512
#define BM 128
#define BN 128
#define BK 32

// ---- gfx1250 async global<->LDS copies (guarded; fallback = VGPR copy) ----
#if defined(__gfx1250__) && __has_builtin(__builtin_amdgcn_global_load_async_to_lds_b128)
#define HAVE_ASYNC_LD 1
#else
#define HAVE_ASYNC_LD 0
#endif
#if defined(__gfx1250__) && __has_builtin(__builtin_amdgcn_global_store_async_from_lds_b128)
#define HAVE_ASYNC_ST 1
#else
#define HAVE_ASYNC_ST 0
#endif

#if HAVE_ASYNC_LD || HAVE_ASYNC_ST
// exact parameter types per hipcc diagnostic:
//   arg0: int __vector(4) addrspace(1)*   (global)
//   arg1: int __vector(4) addrspace(3)*   (LDS)
typedef int vs4i __attribute__((vector_size(16)));
typedef __attribute__((address_space(1))) vs4i gvs4i;
typedef __attribute__((address_space(3))) vs4i lvs4i;
#endif

__device__ __forceinline__ void async_copy16(void* lds, const void* g) {
#if HAVE_ASYNC_LD
  __builtin_amdgcn_global_load_async_to_lds_b128(
      (gvs4i*)const_cast<void*>(g), (lvs4i*)lds, 0, 0);
#else
  *(uint4*)lds = *(const uint4*)g;
#endif
}

__device__ __forceinline__ void async_store16(void* g, const void* lds) {
#if HAVE_ASYNC_ST
  __builtin_amdgcn_global_store_async_from_lds_b128(
      (gvs4i*)g, (lvs4i*)const_cast<void*>(lds), 0, 0);
#else
  *(uint4*)g = *(const uint4*)lds;
#endif
}

__device__ __forceinline__ void async_join() {
#if HAVE_ASYNC_LD || HAVE_ASYNC_ST
#if __has_builtin(__builtin_amdgcn_s_wait_asynccnt)
  __builtin_amdgcn_s_wait_asynccnt(0);
#else
  asm volatile("s_wait_asynccnt 0" ::: "memory");
#endif
#endif
}

__device__ __forceinline__ bf16x16 cat8(bf16x8 a, bf16x8 b) {
  return __builtin_shufflevector(a, b, 0,1,2,3,4,5,6,7,8,9,10,11,12,13,14,15);
}

// window-order row -> original x row
__device__ __forceinline__ int gather_row(int gm) {
  int b   = gm >> 12;          // 4096 rows per batch
  int r   = gm & 4095;
  int win = r >> 4;            // 0..255
  int t   = r & 15;
  int ih  = win >> 4, iw = win & 15;
  int ty  = t >> 2,  tx = t & 3;
  int n   = ((ih << 2) + ty) * 64 + ((iw << 2) + tx);
  return (b << 12) + n;
}

// C[M,512] = A[M,512] @ W[512,512] + bias ; software-pipelined bf16 WMMA core,
// LDS-staged coalesced epilogue.
template<bool GATHER, bool ABF16, bool OUTF32>
__global__ __launch_bounds__(256)
void gemm512_wmma(const void* __restrict__ Av, const float* __restrict__ W,
                  const float* __restrict__ bias, void* __restrict__ Cv) {
  __shared__ __bf16 Alds[BM][40];   // [m][k], 80B rows (16B aligned)
  __shared__ __bf16 Blds[BN][40];   // transposed: [n][k]
  __shared__ alignas(16) unsigned char Craw[BM * (BN + 8) * (OUTF32 ? 4 : 2)];

  const int tid  = threadIdx.x;
  const int lane = tid & 31;
  const int wid  = tid >> 5;
  const int lr   = lane & 15;
  const int hi   = lane >> 4;
  const int waveM = wid & 3;        // 4 waves along M: 32 rows each
  const int waveN = wid >> 2;       // 2 waves along N: 64 cols each
  const int m0 = blockIdx.y * BM;   // N-blocks fastest -> A-tile L2 reuse
  const int n0 = blockIdx.x * BN;

  // A-tile mapping: 2 threads/row, 16 elems each
  const int arow = tid >> 1;
  const int acol = (tid & 1) * 16;
  int src_row = m0 + arow;
  if (GATHER) src_row = gather_row(src_row);
  // B-tile mapping: 2 k-rows x 8 n-cols per thread (row-contiguous loads,
  // packed bf16x2 transposed stores -> ds_store_b32)
  const int brow = (tid >> 4) * 2;  // 0,2,...,30
  const int bcol = (tid & 15) * 8;  // 0..120

  // staging registers (one set dead per instantiation)
  float4 fa0, fa1, fa2, fa3;
  uint4  ua0, ua1;
  float4 gb0, gb1, gb2, gb3;

  auto global_load = [&](int k0) {
    if constexpr (ABF16) {
      const __bf16* ap = (const __bf16*)Av + (size_t)src_row * CDIM + k0 + acol;
      ua0 = *(const uint4*)(ap);
      ua1 = *(const uint4*)(ap + 8);
    } else {
      const float* ap = (const float*)Av + (size_t)src_row * CDIM + k0 + acol;
      fa0 = *(const float4*)(ap);
      fa1 = *(const float4*)(ap + 4);
      fa2 = *(const float4*)(ap + 8);
      fa3 = *(const float4*)(ap + 12);
    }
    const float* bp0 = W + (size_t)(k0 + brow) * CDIM + n0 + bcol;
    gb0 = *(const float4*)(bp0);
    gb1 = *(const float4*)(bp0 + 4);
    gb2 = *(const float4*)(bp0 + CDIM);
    gb3 = *(const float4*)(bp0 + CDIM + 4);
  };

  auto lds_store = [&]() {
    if constexpr (ABF16) {
      *(uint4*)&Alds[arow][acol]     = ua0;
      *(uint4*)&Alds[arow][acol + 8] = ua1;
    } else {
      bf16x8 c0, c1;
      c0[0]=(__bf16)fa0.x; c0[1]=(__bf16)fa0.y; c0[2]=(__bf16)fa0.z; c0[3]=(__bf16)fa0.w;
      c0[4]=(__bf16)fa1.x; c0[5]=(__bf16)fa1.y; c0[6]=(__bf16)fa1.z; c0[7]=(__bf16)fa1.w;
      c1[0]=(__bf16)fa2.x; c1[1]=(__bf16)fa2.y; c1[2]=(__bf16)fa2.z; c1[3]=(__bf16)fa2.w;
      c1[4]=(__bf16)fa3.x; c1[5]=(__bf16)fa3.y; c1[6]=(__bf16)fa3.z; c1[7]=(__bf16)fa3.w;
      *(bf16x8*)&Alds[arow][acol]     = c0;
      *(bf16x8*)&Alds[arow][acol + 8] = c1;
    }
    const float r0[8] = {gb0.x,gb0.y,gb0.z,gb0.w, gb1.x,gb1.y,gb1.z,gb1.w};
    const float r1[8] = {gb2.x,gb2.y,gb2.z,gb2.w, gb3.x,gb3.y,gb3.z,gb3.w};
#pragma unroll
    for (int i = 0; i < 8; ++i) {
      bf16x2 p;                      // (k, k+1) adjacent in [n][k] layout
      p[0] = (__bf16)r0[i];
      p[1] = (__bf16)r1[i];
      *(bf16x2*)&Blds[bcol + i][brow] = p;   // ds_store_b32
    }
  };

  f32x8 acc[2][4] = {};

  global_load(0);                              // prologue
  for (int k0 = 0; k0 < CDIM; k0 += BK) {
    __syncthreads();                           // prior frag reads done
    lds_store();
    __syncthreads();
    if (k0 + BK < CDIM) global_load(k0 + BK);  // overlap with WMMA below
    if (k0 + 2 * BK < CDIM) {                  // deep prefetch -> global_prefetch_b8
      if constexpr (ABF16)
        __builtin_prefetch((const __bf16*)Av + (size_t)src_row * CDIM + k0 + 2*BK + acol, 0, 1);
      else
        __builtin_prefetch((const float*)Av + (size_t)src_row * CDIM + k0 + 2*BK + acol, 0, 1);
      __builtin_prefetch(W + (size_t)(k0 + 2*BK + brow) * CDIM + n0 + bcol, 0, 1);
    }

    bf16x16 af[2], bfr[4];
#pragma unroll
    for (int i = 0; i < 2; ++i) {
      const __bf16* p = &Alds[waveM * 32 + i * 16 + lr][hi * 8];
      af[i] = cat8(*(const bf16x8*)p, *(const bf16x8*)(p + 16));
    }
#pragma unroll
    for (int j = 0; j < 4; ++j) {
      const __bf16* p = &Blds[waveN * 64 + j * 16 + lr][hi * 8];
      bfr[j] = cat8(*(const bf16x8*)p, *(const bf16x8*)(p + 16));
    }
#pragma unroll
    for (int i = 0; i < 2; ++i)
#pragma unroll
      for (int j = 0; j < 4; ++j)
        acc[i][j] = __builtin_amdgcn_wmma_f32_16x16x32_bf16(
            false, af[i], false, bfr[j], (short)0, acc[i][j], false, false);
  }

  // ---- epilogue: bias, stage C tile in LDS, coalesced b128 write-out
  __syncthreads();
#pragma unroll
  for (int j = 0; j < 4; ++j) {
    const int col = waveN * 64 + j * 16 + lr;
    const float bv = bias[n0 + col];
#pragma unroll
    for (int i = 0; i < 2; ++i) {
      const int rowb = waveM * 32 + i * 16 + hi * 8;
#pragma unroll
      for (int r = 0; r < 8; ++r) {
        float vout = acc[i][j][r] + bv;
        if constexpr (OUTF32)
          ((float*)Craw)[(size_t)(rowb + r) * (BN + 8) + col] = vout;
        else
          ((__bf16*)Craw)[(size_t)(rowb + r) * (BN + 8) + col] = (__bf16)vout;
      }
    }
  }
  __syncthreads();
  {
    const int orow = tid >> 1;
    const int oh   = (tid & 1) * 64;           // 64-element half-rows
    if constexpr (OUTF32) {
      const float* src = (const float*)Craw + (size_t)orow * (BN + 8) + oh;
      float* dst = (float*)Cv + (size_t)(m0 + orow) * CDIM + n0 + oh;
#pragma unroll
      for (int u = 0; u < 16; ++u) ((float4*)dst)[u] = ((const float4*)src)[u];
    } else {
      const __bf16* src = (const __bf16*)Craw + (size_t)orow * (BN + 8) + oh;
      __bf16* dst = (__bf16*)Cv + (size_t)(m0 + orow) * CDIM + n0 + oh;
#pragma unroll
      for (int u = 0; u < 8; ++u) ((uint4*)dst)[u] = ((const uint4*)src)[u];
    }
  }
}

// One block per window (16 contiguous rows), one wave per head.
// Q/K/V/ctx staged through LDS via async global<->LDS copies (ASYNCcnt path).
__global__ __launch_bounds__(256)
void win_attn_wmma(const __bf16* __restrict__ q, const __bf16* __restrict__ k,
                   const __bf16* __restrict__ v, __bf16* __restrict__ ctx) {
  __shared__ __bf16 Qlds[16][520];
  __shared__ __bf16 Klds[16][520];
  __shared__ __bf16 Vlds[16][520];
  __shared__ __bf16 Clds[16][520];
  __shared__ __bf16 Plds[8][16][24];  // probs per head, 16B-aligned rows

  const int tid  = threadIdx.x;
  const int lane = tid & 31;
  const int wid  = tid >> 5;          // head 0..7
  const int lr   = lane & 15;
  const int hi   = lane >> 4;
  const int rowbase = blockIdx.x * 16;
  const int colbase = wid * 64;

  // ---- stage whole-window Q/K/V (16x512 each): async b128 copies, no VGPR trip
  {
    const int srow = tid >> 4;          // 16 threads per row
    const int scol = (tid & 15) * 32;   // 32 elems each
    const size_t goff = (size_t)(rowbase + srow) * CDIM + scol;
#pragma unroll
    for (int u = 0; u < 4; ++u) {
      async_copy16(&Qlds[srow][scol + u * 8], q + goff + u * 8);
      async_copy16(&Klds[srow][scol + u * 8], k + goff + u * 8);
      async_copy16(&Vlds[srow][scol + u * 8], v + goff + u * 8);
    }
    async_join();                       // s_wait_asynccnt 0 before barrier
  }
  __syncthreads();

  // ---- S = Q @ K^T : B-fragment of K^T == contiguous rows of K
  f32x8 s = {};
#pragma unroll
  for (int ks = 0; ks < 64; ks += 32) {
    const __bf16* qp = &Qlds[lr][colbase + ks + hi * 8];
    bf16x16 af = cat8(*(const bf16x8*)qp, *(const bf16x8*)(qp + 16));
    const __bf16* kp = &Klds[lr][colbase + ks + hi * 8];
    bf16x16 bf = cat8(*(const bf16x8*)kp, *(const bf16x8*)(kp + 16));
    s = __builtin_amdgcn_wmma_f32_16x16x32_bf16(false, af, false, bf, (short)0, s, false, false);
  }

  // ---- sigmoid(S/8) via fast rcp (result truncated to bf16 anyway)
#pragma unroll
  for (int r = 0; r < 8; ++r) {
    float p = __builtin_amdgcn_rcpf(1.0f + __expf(-0.125f * s[r]));
    Plds[wid][hi * 8 + r][lr] = (__bf16)p;
  }
  __syncthreads();

  // ---- ctx = P @ V  (K padded 16 -> 32 with zeros)
  const bf16x8 z8 = { (__bf16)0.f,(__bf16)0.f,(__bf16)0.f,(__bf16)0.f,
                      (__bf16)0.f,(__bf16)0.f,(__bf16)0.f,(__bf16)0.f };
  bf16x16 pa = cat8(*(const bf16x8*)&Plds[wid][lr][hi * 8], z8);

#pragma unroll
  for (int j = 0; j < 4; ++j) {
    bf16x16 bv = cat8(z8, z8);
    {                                   // elements 8..15 carry the zero-padded K=16..31
#pragma unroll
      for (int kk = 0; kk < 8; ++kk) bv[kk] = Vlds[hi * 8 + kk][colbase + j * 16 + lr];
    }
    f32x8 c = {};
    c = __builtin_amdgcn_wmma_f32_16x16x32_bf16(false, pa, false, bv, (short)0, c, false, false);
#pragma unroll
    for (int r = 0; r < 8; ++r)
      Clds[hi * 8 + r][colbase + j * 16 + lr] = (__bf16)c[r];
  }
  __syncthreads();

  // ---- ctx write-out: async LDS->global b128 stores
  {
    const int srow = tid >> 4;
    const int scol = (tid & 15) * 32;
    const size_t goff = (size_t)(rowbase + srow) * CDIM + scol;
#pragma unroll
    for (int u = 0; u < 4; ++u)
      async_store16(ctx + goff + u * 8, &Clds[srow][scol + u * 8]);
    async_join();
  }
}

extern "C" void kernel_launch(void* const* d_in, const int* in_sizes, int n_in,
                              void* d_out, int out_size, void* d_ws, size_t ws_size,
                              hipStream_t stream) {
  const float* x  = (const float*)d_in[0];
  const float* Wq = (const float*)d_in[1];
  const float* bq = (const float*)d_in[2];
  const float* Wk = (const float*)d_in[3];
  const float* bk = (const float*)d_in[4];
  const float* Wv = (const float*)d_in[5];
  const float* bv = (const float*)d_in[6];
  const float* Wo = (const float*)d_in[7];
  const float* bo = (const float*)d_in[8];
  float* out = (float*)d_out;

  const size_t M = 32 * 4096;           // rows (window-ordered)
  __bf16* qb   = (__bf16*)d_ws;
  __bf16* kb   = qb + M * CDIM;
  __bf16* vbuf = kb + M * CDIM;
  __bf16* cb   = vbuf + M * CDIM;

  dim3 g(CDIM / BN, M / BM);            // N-blocks fastest -> A-tile L2 reuse
  dim3 b(256);

  // Q/K/V projections: gather x rows into window order, fp32 in, bf16 out
  gemm512_wmma<true,  false, false><<<g, b, 0, stream>>>(x, Wq, bq, qb);
  gemm512_wmma<true,  false, false><<<g, b, 0, stream>>>(x, Wk, bk, kb);
  gemm512_wmma<true,  false, false><<<g, b, 0, stream>>>(x, Wv, bv, vbuf);

  // Per-window sigmoid attention (8192 windows x 8 heads)
  win_attn_wmma<<<dim3(M / 16), b, 0, stream>>>(qb, kb, vbuf, cb);

  // Output projection: bf16 ctx in, fp32 out (already in reference's window order)
  gemm512_wmma<false, true,  true ><<<g, b, 0, stream>>>(cb, Wo, bo, out);
}
